// Encoding_86595130622077
// MI455X (gfx1250) — matrix operand order repaired and checked by reading, not compiled, not run on hardware
//
#include <hip/hip_runtime.h>
#include <stdint.h>

#define Bn 32
#define Nn 1024
#define Dn 128
#define TI 128
#define TJ 128
#define NT (Nn / TJ)
#define LSTR 136          // LDS row stride (bf16 elems): 128 data + 8 pad (=272B rows)
#define WSTR 264          // LDS row stride for 256-wide transposed weights

// LDS layout (bytes)
#define OFF_Q    0u
#define OFF_PT   34816u
#define OFF_SP   69632u
#define OFF_CB   104448u
#define OFF_K0   104960u
#define OFF_VT0  139776u
#define OFF_K1   174592u
#define OFF_VT1  209408u
#define OFF_WT   104960u            // overlays K0+VT0 in epilogue
#define SMEM_BYTES 244224u

typedef __attribute__((ext_vector_type(16))) __bf16 v16bf;
typedef __attribute__((ext_vector_type(8)))  float  v8f;
typedef __attribute__((ext_vector_type(4)))  unsigned int u32x4;
typedef __attribute__((ext_vector_type(4)))  int v4i;
typedef __attribute__((ext_vector_type(8)))  int v8i;

union FragU { u32x4 q[2]; v16bf v; };

__device__ __forceinline__ unsigned short f2bf(float f) {
  unsigned int u = __builtin_bit_cast(unsigned int, f);
  u += 0x7FFFu + ((u >> 16) & 1u);                 // round-to-nearest-even
  return (unsigned short)(u >> 16);
}

// A-operand (16x32 bf16): lane m = lane&15; lanes 0-15 K=0..7,16..23; lanes 16-31 K=8..15,24..31
__device__ __forceinline__ v16bf ldA(const unsigned short* row, int kbase, int hi) {
  FragU f;
  const int o0 = kbase + hi * 8;
  f.q[0] = *(const u32x4*)(row + o0);
  f.q[1] = *(const u32x4*)(row + o0 + 16);
  return f.v;
}

// B-operand (32x16 bf16): lane n = lane&15; lanes 0-15 K=0..7,16..23; lanes 16-31 K=8..15,24..31
__device__ __forceinline__ v16bf ldB(const unsigned short* row, int kbase, int hi) {
  FragU f;
  const int o0 = kbase + hi * 8;
  f.q[0] = *(const u32x4*)(row + o0);
  f.q[1] = *(const u32x4*)(row + o0 + 16);
  return f.v;
}

__device__ __forceinline__ v8f wmma_bf16(v16bf a, v16bf b, v8f c) {
  return __builtin_amdgcn_wmma_f32_16x16x32_bf16(false, a, false, b, (short)0, c, false, false);
}

// ---- Tensor Data Mover: 2D 128x128 bf16 tile, global -> LDS, padded rows (64DW + 4DW pad) ----
__device__ __forceinline__ void tdm_load_tile(unsigned lds_off, const unsigned short* g,
                                              unsigned row_stride_elems) {
  unsigned long long ga = (unsigned long long)(uintptr_t)g;
  u32x4 g0;
  g0.x = 1u;                                              // count=1, user descriptor
  g0.y = lds_off;                                         // LDS byte address
  g0.z = (unsigned)ga;                                    // global_addr[31:0]
  g0.w = (unsigned)((ga >> 32) & 0x01FFFFFFull) | (2u << 30);  // global_addr[56:32] | type=2
  v8i g1;
  g1[0] = (int)((1u << 16)      // data_size = 2 bytes
              | (1u << 20)      // pad_enable
              | (5u << 22)      // pad_interval: 64 DWORDs (=256B row)
              | (3u << 25));    // pad_amount:   4 DWORDs (=16B  -> 272B row stride)
  g1[1] = (int)((128u & 0xFFFFu) << 16);                  // tensor_dim0[15:0]=128 (bits 63:48)
  g1[2] = (int)(((128u >> 16) & 0xFFFFu) | ((128u & 0xFFFFu) << 16)); // dim0 hi | dim1 lo
  g1[3] = (int)(((128u >> 16) & 0xFFFFu) | (128u << 16)); // dim1 hi | tile_dim0=128
  g1[4] = 128;                                            // tile_dim1=128, tile_dim2=0
  g1[5] = (int)row_stride_elems;                          // tensor_dim0_stride[31:0]
  g1[6] = 0;
  g1[7] = 0;
  v4i z4 = {0, 0, 0, 0};
#if defined(__clang_major__) && __clang_major__ >= 23
  v8i z8 = {0, 0, 0, 0, 0, 0, 0, 0};
  __builtin_amdgcn_tensor_load_to_lds(g0, g1, z4, z4, z8, 0);
#else
  __builtin_amdgcn_tensor_load_to_lds(g0, g1, z4, z4, 0);
#endif
}

// ---------------- pre-pass 1: cb[b,j] = P[b,j,:] . wa  (wa = w_itr_att[0:128]) ----------------
__global__ __launch_bounds__(256)
void colbias_kernel(const float* __restrict__ P, const float* __restrict__ watt,
                    float* __restrict__ cb) {
  const int row  = blockIdx.x * 8 + (threadIdx.x >> 5);
  const int lane = threadIdx.x & 31;
  const float* p = P + (size_t)row * Dn;
  float s = 0.f;
  #pragma unroll
  for (int c = 0; c < Dn; c += 32) s += p[c + lane] * watt[c + lane];
  s += __shfl_xor(s, 16, 32);
  s += __shfl_xor(s,  8, 32);
  s += __shfl_xor(s,  4, 32);
  s += __shfl_xor(s,  2, 32);
  s += __shfl_xor(s,  1, 32);
  if (lane == 0) cb[row] = s;
}

// ------ pre-pass 2: Pbf = bf16(P) row-major; PbfT[b][d][n] = bf16(P[b][n][d]) ------
__global__ __launch_bounds__(256)
void cvt_kernel(const float* __restrict__ P, unsigned short* __restrict__ Pbf,
                unsigned short* __restrict__ PbfT) {
  __shared__ unsigned short tile[64][72];
  const int blk = blockIdx.x;           // Bn * (Nn/64) * (Dn/64) = 1024 blocks
  const int b   = blk >> 5;
  const int t2  = blk & 31;
  const int n0  = (t2 >> 1) * 64;
  const int d0  = (t2 & 1) * 64;
  const int tr  = threadIdx.x >> 6;     // 0..3
  const int tc  = threadIdx.x & 63;     // 0..63
  #pragma unroll
  for (int k = 0; k < 16; ++k) {
    const int r = tr + k * 4;           // n-row within tile
    unsigned short v = f2bf(P[((size_t)b * Nn + n0 + r) * Dn + d0 + tc]);
    Pbf[((size_t)b * Nn + n0 + r) * Dn + d0 + tc] = v;
    tile[tc][r] = v;                    // transposed position [d][n]
  }
  __syncthreads();
  #pragma unroll
  for (int k = 0; k < 16; ++k) {
    const int dr = tr + k * 4;          // d-row within tile
    PbfT[((size_t)b * Dn + d0 + dr) * Nn + n0 + tc] = tile[dr][tc];
  }
}

// ---------------- fused flash-attention + gated MLP ----------------
__global__ __launch_bounds__(256)
void fused_attn_mlp(const float* __restrict__ P,
                    const float* __restrict__ cb,
                    const float* __restrict__ watt,
                    const unsigned short* __restrict__ Pbf,
                    const unsigned short* __restrict__ PbfT,
                    const float* __restrict__ w1,
                    const float* __restrict__ w2,
                    const float* __restrict__ w3,
                    const float* __restrict__ b1,
                    const float* __restrict__ b2,
                    const float* __restrict__ b3,
                    float* __restrict__ out) {
  extern __shared__ char smem[];
  unsigned short* sQ  = (unsigned short*)(smem + OFF_Q);
  unsigned short* sPt = (unsigned short*)(smem + OFF_PT);
  unsigned short* sSP = (unsigned short*)(smem + OFF_SP);
  float*          sCb = (float*)(smem + OFF_CB);
  unsigned short* sWT = (unsigned short*)(smem + OFF_WT);

  const int b    = blockIdx.x >> 3;
  const int i0   = (blockIdx.x & 7) * TI;
  const int tid  = threadIdx.x;
  const int wave = tid >> 5;
  const int lane = tid & 31;
  const int lo16 = lane & 15;
  const int hi   = lane >> 4;

  const float* Pb  = P  + (size_t)b * Nn * Dn;
  const float* cbB = cb + (size_t)b * Nn;
  const unsigned short* PbfB  = Pbf  + (size_t)b * Nn * Dn;
  const unsigned short* PbfTB = PbfT + (size_t)b * Dn * Nn;

  // kick off TDM for tile 0 (K row-major from Pbf, V^T from PbfT), wave 0 only
  if (wave == 0) {
    tdm_load_tile(OFF_K0,  PbfB, 128u);
    tdm_load_tile(OFF_VT0, PbfTB, 1024u);
  }

  // ---- stage Q = P*wc and Pt (bf16) ----
  {
    const int r  = tid >> 1;
    const int c0 = (tid & 1) * 64;
    const float* src = Pb + (size_t)(i0 + r) * Dn + c0;
    const float* wcp = watt + 2 * Dn + c0;
    unsigned short* qrow = sQ  + r * LSTR + c0;
    unsigned short* prow = sPt + r * LSTR + c0;
    #pragma unroll
    for (int c = 0; c < 64; c += 4) {
      float4 v  = *(const float4*)(src + c);
      float4 wv = *(const float4*)(wcp + c);
      prow[c + 0] = f2bf(v.x); prow[c + 1] = f2bf(v.y);
      prow[c + 2] = f2bf(v.z); prow[c + 3] = f2bf(v.w);
      qrow[c + 0] = f2bf(v.x * wv.x); qrow[c + 1] = f2bf(v.y * wv.y);
      qrow[c + 2] = f2bf(v.z * wv.z); qrow[c + 3] = f2bf(v.w * wv.w);
    }
  }

  float mrun[8], lrun[8];
  v8f accO[8];
  const v8f vz = {0.f, 0.f, 0.f, 0.f, 0.f, 0.f, 0.f, 0.f};
  #pragma unroll
  for (int r = 0; r < 8; ++r) { mrun[r] = -1e30f; lrun[r] = 0.f; }
  #pragma unroll
  for (int t = 0; t < 8; ++t) accO[t] = vz;

  unsigned short* sp = sSP + wave * 16 * LSTR;   // wave-private 16xLSTR strip

  #pragma unroll 1
  for (int jt = 0; jt < NT; ++jt) {
    const int j0 = jt * TJ;
    __syncthreads();  // everyone done with the buffer the next TDM will overwrite
    if (wave == 0) {
      if (jt + 1 < NT) {
        const unsigned koff  = ((jt + 1) & 1) ? OFF_K1  : OFF_K0;
        const unsigned vtoff = ((jt + 1) & 1) ? OFF_VT1 : OFF_VT0;
        tdm_load_tile(koff,  PbfB + (size_t)(j0 + TJ) * Dn, 128u);
        tdm_load_tile(vtoff, PbfTB + (j0 + TJ), 1024u);
        __builtin_amdgcn_s_wait_tensorcnt(2);  // current tile's pair complete
      } else {
        __builtin_amdgcn_s_wait_tensorcnt(0);
      }
    }
    if (tid < TJ) sCb[tid] = cbB[j0 + tid];
    __syncthreads();  // tile jt visible to all waves

    const unsigned short* kb = (const unsigned short*)(smem + ((jt & 1) ? OFF_K1  : OFF_K0));
    const unsigned short* vt = (const unsigned short*)(smem + ((jt & 1) ? OFF_VT1 : OFF_VT0));

    // ---- S = (P*wc) @ K^T  (16x128 per wave), B-fragments pipelined ----
    v8f accS[8];
    #pragma unroll
    for (int t = 0; t < 8; ++t) accS[t] = vz;
    #pragma unroll
    for (int kc = 0; kc < 4; ++kc) {
      v16bf af = ldA(sQ + (wave * 16 + lo16) * LSTR, kc * 32, hi);
      v16bf bcur = ldB(kb + lo16 * LSTR, kc * 32, hi);
      #pragma unroll
      for (int t = 0; t < 8; ++t) {
        v16bf bnxt = bcur;
        if (t < 7) bnxt = ldB(kb + ((t + 1) * 16 + lo16) * LSTR, kc * 32, hi);
        accS[t] = wmma_bf16(af, bcur, accS[t]);
        bcur = bnxt;
      }
    }
    // column bias cb[j] (the row-bias term cancels in softmax)
    #pragma unroll
    for (int t = 0; t < 8; ++t) accS[t] += sCb[t * 16 + lo16];

    // ---- online softmax (rows replicated across 16-lane halves) ----
    float sc[8];
    #pragma unroll
    for (int r = 0; r < 8; ++r) {
      float pm = accS[0][r];
      #pragma unroll
      for (int t = 1; t < 8; ++t) pm = fmaxf(pm, accS[t][r]);
      pm = fmaxf(pm, __shfl_xor(pm, 1, 32));
      pm = fmaxf(pm, __shfl_xor(pm, 2, 32));
      pm = fmaxf(pm, __shfl_xor(pm, 4, 32));
      pm = fmaxf(pm, __shfl_xor(pm, 8, 32));
      float mnew = fmaxf(mrun[r], pm);
      sc[r] = __expf(mrun[r] - mnew);
      mrun[r] = mnew;
      lrun[r] *= sc[r];
    }
    float ps[8];
    #pragma unroll
    for (int r = 0; r < 8; ++r) ps[r] = 0.f;
    #pragma unroll
    for (int t = 0; t < 8; ++t) {
      #pragma unroll
      for (int r = 0; r < 8; ++r) {
        accO[t][r] *= sc[r];
        float p = __expf(accS[t][r] - mrun[r]);
        accS[t][r] = p;
        ps[r] += p;
      }
    }
    #pragma unroll
    for (int r = 0; r < 8; ++r) {
      float s = ps[r];
      s += __shfl_xor(s, 1, 32);
      s += __shfl_xor(s, 2, 32);
      s += __shfl_xor(s, 4, 32);
      s += __shfl_xor(s, 8, 32);
      lrun[r] += s;
    }

    // ---- bounce probabilities through wave-private LDS to get A-layout ----
    #pragma unroll
    for (int t = 0; t < 8; ++t)
      #pragma unroll
      for (int r = 0; r < 8; ++r)
        sp[(r + 8 * hi) * LSTR + t * 16 + lo16] = f2bf(accS[t][r]);

    // ---- O += Pij @ V (B-fragments from V^T tile, pipelined) ----
    #pragma unroll
    for (int kc = 0; kc < 4; ++kc) {
      v16bf af = ldA(sp + lo16 * LSTR, kc * 32, hi);
      v16bf bcur = ldB(vt + lo16 * LSTR, kc * 32, hi);
      #pragma unroll
      for (int t = 0; t < 8; ++t) {
        v16bf bnxt = bcur;
        if (t < 7) bnxt = ldB(vt + ((t + 1) * 16 + lo16) * LSTR, kc * 32, hi);
        accO[t] = wmma_bf16(af, bcur, accO[t]);
        bcur = bnxt;
      }
    }
  }

  // ---- normalize O and park it (bf16) in the wave strip for the concat matmul ----
  float inv[8];
  #pragma unroll
  for (int r = 0; r < 8; ++r) inv[r] = 1.0f / lrun[r];
  #pragma unroll
  for (int t = 0; t < 8; ++t)
    #pragma unroll
    for (int r = 0; r < 8; ++r)
      sp[(r + 8 * hi) * LSTR + t * 16 + lo16] = f2bf(accO[t][r] * inv[r]);

  // ---- epilogue: [P | O] (16x256) @ w{1,2,3} (256x128), gates, combine ----
  auto stageW = [&](const float* __restrict__ w) {
    __syncthreads();
    #pragma unroll 4
    for (int idx = tid; idx < 2 * Dn * Dn; idx += 256) {
      int k = idx >> 7;                  // input dim 0..255
      int n = idx & 127;                 // output dim 0..127
      sWT[n * WSTR + k] = f2bf(w[idx]);  // store transposed
    }
    __syncthreads();
  };
  auto gateMM = [&](const float* __restrict__ bias, v8f* acc) {
    #pragma unroll
    for (int t = 0; t < 8; ++t) acc[t] = vz;
    #pragma unroll
    for (int kc = 0; kc < 8; ++kc) {
      const unsigned short* arow = (kc < 4) ? (sPt + (wave * 16 + lo16) * LSTR)
                                            : (sp + lo16 * LSTR);
      v16bf af = ldA(arow, (kc & 3) * 32, hi);
      v16bf bcur = ldB(sWT + lo16 * WSTR, kc * 32, hi);
      #pragma unroll
      for (int t = 0; t < 8; ++t) {
        v16bf bnxt = bcur;
        if (t < 7) bnxt = ldB(sWT + ((t + 1) * 16 + lo16) * WSTR, kc * 32, hi);
        acc[t] = wmma_bf16(af, bcur, acc[t]);
        bcur = bnxt;
      }
    }
    #pragma unroll
    for (int t = 0; t < 8; ++t) acc[t] += bias[t * 16 + lo16];
  };

  v8f zf[8], acc[8];

  // z = tanh(Pc @ w1 + b1)
  stageW(w1); gateMM(b1, acc);
  #pragma unroll
  for (int t = 0; t < 8; ++t)
    #pragma unroll
    for (int r = 0; r < 8; ++r)
      zf[t][r] = tanhf(acc[t][r]);

  // f = sigmoid(Pc @ w3 + b3); zf <- f*z
  stageW(w3); gateMM(b3, acc);
  #pragma unroll
  for (int t = 0; t < 8; ++t)
    #pragma unroll
    for (int r = 0; r < 8; ++r)
      zf[t][r] *= 1.0f / (1.0f + __expf(-acc[t][r]));

  // r = sigmoid(Pc @ w2 + b2); out = r*P + f*z
  stageW(w2); gateMM(b2, acc);
  #pragma unroll
  for (int t = 0; t < 8; ++t) {
    #pragma unroll
    for (int r = 0; r < 8; ++r) {
      const int row = i0 + wave * 16 + r + 8 * hi;
      const int col = t * 16 + lo16;
      float pv = Pb[(size_t)row * Dn + col];
      float rg = 1.0f / (1.0f + __expf(-acc[t][r]));
      out[((size_t)b * Nn + row) * Dn + col] = rg * pv + zf[t][r];
    }
  }
}

extern "C" void kernel_launch(void* const* d_in, const int* in_sizes, int n_in,
                              void* d_out, int out_size, void* d_ws, size_t ws_size,
                              hipStream_t stream) {
  (void)in_sizes; (void)n_in; (void)out_size; (void)ws_size;
  const float* P    = (const float*)d_in[0];
  const float* watt = (const float*)d_in[1];
  const float* w1   = (const float*)d_in[2];
  const float* w2   = (const float*)d_in[3];
  const float* w3   = (const float*)d_in[4];
  const float* b1   = (const float*)d_in[5];
  const float* b2   = (const float*)d_in[6];
  const float* b3   = (const float*)d_in[7];
  float* out = (float*)d_out;

  // workspace layout: cb (128KB) | Pbf (8MB) | PbfT (8MB)
  float*          cbw  = (float*)d_ws;
  unsigned short* Pbf  = (unsigned short*)((char*)d_ws + 131072);
  unsigned short* PbfT = (unsigned short*)((char*)d_ws + 131072 + (size_t)Bn * Nn * Dn * 2);

  cvt_kernel<<<Bn * (Nn / 64) * (Dn / 64), 256, 0, stream>>>(P, Pbf, PbfT);
  colbias_kernel<<<(Bn * Nn) / 8, 256, 0, stream>>>(P, watt, cbw);

  (void)hipFuncSetAttribute((const void*)fused_attn_mlp,
                            hipFuncAttributeMaxDynamicSharedMemorySize, (int)SMEM_BYTES);
  fused_attn_mlp<<<Bn * (Nn / TI), 256, SMEM_BYTES, stream>>>(
      P, cbw, watt, Pbf, PbfT, w1, w2, w3, b1, b2, b3, out);
}
